// InteractionBlock_7945689497812
// MI455X (gfx1250) — matrix operand order repaired and checked by reading, not compiled, not run on hardware
//
#include <hip/hip_runtime.h>
#include <cstdint>
#include <cstddef>

// ============================================================================
// ViT-Adapter InteractionBlock for MI455X (gfx1250, wave32, WMMA).
// GEMMs: bf16 operands, f32 accumulate via v_wmma_f32_16x16x32_bf16.
// Per-block the 64 weight columns are staged into LDS via the gfx1250 async
// global->LDS copy path (ASYNCcnt + double buffering); each wave computes a
// 32x64 tile (8 WMMA per 32-wide k-step) reading B from LDS, A from global.
// ============================================================================

typedef __attribute__((ext_vector_type(16))) __bf16 bf16x16;
typedef __attribute__((ext_vector_type(8)))  __bf16 bf16x8;
typedef __attribute__((ext_vector_type(8)))  float  floatx8;
typedef __attribute__((ext_vector_type(4)))  float  floatx4;
typedef __attribute__((ext_vector_type(4)))  unsigned int uintx4;
typedef __attribute__((ext_vector_type(4)))  int    v4i;
typedef __attribute__((address_space(3)))    v4i    as3_v4i;

#define CDIM 768
#define NHEAD 6
#define NPTS 4
#define DHEAD 128
#define NB 4
#define NVIT 1024
#define NPYR 5376

// ---- async global->LDS staging (gfx1250) ----
#if defined(__has_builtin)
#if __has_builtin(__builtin_amdgcn_global_load_async_to_lds_b128) && \
    __has_builtin(__builtin_amdgcn_s_wait_asynccnt)
#define USE_ASYNC_LDS 1
#endif
#endif

__device__ __forceinline__ void async_copy_b128(const unsigned short* g, unsigned short* l) {
#ifdef USE_ASYNC_LDS
  __builtin_amdgcn_global_load_async_to_lds_b128(
      (v4i*)g,                                  // global source (generic ptr per diag)
      (as3_v4i*)(uintptr_t)l,                   // LDS dest: low 32 bits = LDS offset
      0, 0);
#else
  *reinterpret_cast<uintx4*>(l) = *reinterpret_cast<const uintx4*>(g);
#endif
}

__device__ __forceinline__ void wait_async_le3() {
#ifdef USE_ASYNC_LDS
  __builtin_amdgcn_s_wait_asynccnt(3);
#endif
}
__device__ __forceinline__ void wait_async_le0() {
#ifdef USE_ASYNC_LDS
  __builtin_amdgcn_s_wait_asynccnt(0);
#endif
}

__device__ __forceinline__ unsigned short f2bf(float f) {
  unsigned int u = __builtin_bit_cast(unsigned int, f);
  u += 0x7fffu + ((u >> 16) & 1u);   // round-to-nearest-even
  return (unsigned short)(u >> 16);
}

__device__ __forceinline__ bf16x8 load_bf8(const unsigned short* p) {
  uintx4 u = *reinterpret_cast<const uintx4*>(p);
  return __builtin_bit_cast(bf16x8, u);
}

__device__ __forceinline__ bf16x16 combine16(bf16x8 lo, bf16x8 hi) {
  return __builtin_shufflevector(lo, hi,
      0, 1, 2, 3, 4, 5, 6, 7, 8, 9, 10, 11, 12, 13, 14, 15);
}

// ---------------------------------------------------------------------------
// Weight pack: f32 W (K,N) row-major -> bf16 Wt (Npad,K) row-major
// (transposed, columns >= N zero-filled).
// ---------------------------------------------------------------------------
__global__ void packw_kernel(const float* __restrict__ W, unsigned short* __restrict__ Wt,
                             int K, int N, int Npad) {
  int i = blockIdx.x * 256 + threadIdx.x;     // index over (Npad, K) output
  if (i >= Npad * K) return;
  int n = i / K, k = i % K;
  Wt[i] = (n < N) ? f2bf(W[(size_t)k * N + n]) : (unsigned short)0;
}

// ---------------------------------------------------------------------------
// LayerNorm (eps 1e-6) over C=768, output bf16. One block (256 thr) per token.
// ---------------------------------------------------------------------------
__global__ void ln_bf16_kernel(const float* __restrict__ X, const float* __restrict__ g,
                               const float* __restrict__ bt, unsigned short* __restrict__ O) {
  int t = blockIdx.x;
  int tid = threadIdx.x;
  const float* x = X + (size_t)t * CDIM;
  float v0 = x[tid], v1 = x[tid + 256], v2 = x[tid + 512];
  __shared__ float s1[256], s2[256];
  s1[tid] = v0 + v1 + v2;
  s2[tid] = v0 * v0 + v1 * v1 + v2 * v2;
  __syncthreads();
  for (int off = 128; off > 0; off >>= 1) {
    if (tid < off) { s1[tid] += s1[tid + off]; s2[tid] += s2[tid + off]; }
    __syncthreads();
  }
  float mu = s1[0] * (1.0f / CDIM);
  float var = s2[0] * (1.0f / CDIM) - mu * mu;
  float r = rsqrtf(var + 1e-6f);
  unsigned short* o = O + (size_t)t * CDIM;
  o[tid]       = f2bf((v0 - mu) * r * g[tid]       + bt[tid]);
  o[tid + 256] = f2bf((v1 - mu) * r * g[tid + 256] + bt[tid + 256]);
  o[tid + 512] = f2bf((v2 - mu) * r * g[tid + 512] + bt[tid + 512]);
}

// ---------------------------------------------------------------------------
// GEMM: Out[M,Npad] (f32) = A[M,K](bf16) @ Wt[Npad,K](bf16)^T + bias[:N]
// mode 0: plain   mode 1: res + gamma[n]*val   mode 2: res + val
// Wave computes 32x64 (2 M-tiles x 4 N-tiles); block = 8 waves along M; all
// waves in a block share n0, so the 64 B-columns are staged to LDS once,
// double-buffered in KCHUNK=96-half chunks via async copies.
// Requires: M % 256 == 0, Npad % 64 == 0, K % 96 == 0. Wt zero-padded.
// Fragment layouts follow CDNA5 ISA 7.12.2 (wave32).
// ---------------------------------------------------------------------------
#define KCHUNK 96        // halfs of K staged per chunk
#define PADK   104       // LDS row stride in halfs (52 dwords -> conflict-free)

__global__ void gemm_bf16_wmma_kernel(const unsigned short* __restrict__ A,
                                      const unsigned short* __restrict__ Wt,
                                      const float* __restrict__ bias,
                                      const float* __restrict__ res,
                                      const float* __restrict__ gamma,
                                      float* __restrict__ Out,
                                      int Npad, int Nvalid, int K, int mode) {
  __shared__ unsigned short bsh[2][64 * PADK];
  int tid  = threadIdx.x;
  int wave = tid >> 5;
  int lane = tid & 31;
  int m0 = (blockIdx.y * 8 + wave) * 32;
  int n0 = blockIdx.x * 64;
  int lhalf = lane >> 4;                        // 0: lanes 0-15, 1: lanes 16-31
  int l15   = lane & 15;
  // A: lane holds row; halves 0-7 = K(k + lhalf*8), halves 8-15 = K(k+16+lhalf*8)
  const unsigned short* arow0 = A + (size_t)(m0 + l15) * K + lhalf * 8;
  const unsigned short* arow1 = arow0 + (size_t)16 * K;
  const unsigned short* wbase = Wt + (size_t)n0 * K;   // 64 rows x K

  // 64 rows x 96 halfs = 768 x 16B transfers per chunk = 3 per thread
  auto issue_chunk = [&](int kc, int buf) {
#pragma unroll
    for (int t = 0; t < 3; t++) {
      int i = tid + t * 256;                    // 0..767
      int row = i / 12, seg = i % 12;
      const unsigned short* g = wbase + (size_t)row * K + kc * KCHUNK + seg * 8;
      unsigned short* l = &bsh[buf][row * PADK + seg * 8];
      async_copy_b128(g, l);
    }
  };

  int nch = K / KCHUNK;
  issue_chunk(0, 0);

  floatx8 acc[8] = {};
  for (int kc = 0; kc < nch; kc++) {
    int buf = kc & 1;
    __syncthreads();                            // all waves done reading buf^1
    if (kc + 1 < nch) { issue_chunk(kc + 1, buf ^ 1); wait_async_le3(); }
    else              { wait_async_le0(); }
    __syncthreads();                            // chunk kc visible to all waves
    int kbase = kc * KCHUNK;
#pragma unroll
    for (int kk = 0; kk < KCHUNK; kk += 32) {
      int k = kbase + kk;
      bf16x16 af0 = combine16(load_bf8(arow0 + k), load_bf8(arow0 + k + 16));
      bf16x16 af1 = combine16(load_bf8(arow1 + k), load_bf8(arow1 + k + 16));
#pragma unroll
      for (int j = 0; j < 4; j++) {
        const unsigned short* bp = &bsh[buf][(l15 + j * 16) * PADK + kk + lhalf * 8];
        bf16x16 bf = combine16(load_bf8(bp), load_bf8(bp + 16));
        acc[j]     = __builtin_amdgcn_wmma_f32_16x16x32_bf16(
            false, af0, false, bf, (short)0, acc[j], false, false);
        acc[j + 4] = __builtin_amdgcn_wmma_f32_16x16x32_bf16(
            false, af1, false, bf, (short)0, acc[j + 4], false, false);
      }
    }
  }

#pragma unroll
  for (int j = 0; j < 4; j++) {
    int n = n0 + j * 16 + l15;
    if (n >= Nvalid) continue;                  // cold guard, padded cols never read
    float bi = bias[n];
    float ga = (mode == 1) ? gamma[n] : 0.0f;
#pragma unroll
    for (int half = 0; half < 2; half++) {
      floatx8 a = half ? acc[j + 4] : acc[j];
      int mb = m0 + half * 16 + (lhalf << 3);   // D layout: lanes 16-31 hold M = 8+i
#pragma unroll
      for (int i = 0; i < 8; i++) {
        size_t o = (size_t)(mb + i) * Npad + n;
        float v = a[i] + bi;
        if (mode == 1)      v = res[o] + ga * v;
        else if (mode == 2) v = res[o] + v;
        Out[o] = v;
      }
    }
  }
}

// ---------------------------------------------------------------------------
// Multi-scale deformable attention core.
// CFG 0 (injector): queries = 32x32 ViT grid, sample pyramid (64^2,32^2,16^2)
// CFG 1 (extractor): queries = pyramid tokens, sample 32x32 ViT grid
// One wave per (b, q, head); lane covers 4 of Dh=128 channels. Output bf16.
// OFF/AW buffers use the padded GEMM row strides.
// ---------------------------------------------------------------------------
template <int CFG>
__global__ void msda_kernel(const float* __restrict__ V, const float* __restrict__ OFF,
                            const float* __restrict__ AW, unsigned short* __restrict__ O) {
  constexpr int L      = (CFG == 0) ? 3 : 1;
  constexpr int Nq     = (CFG == 0) ? NVIT : NPYR;
  constexpr int S      = (CFG == 0) ? NPYR : NVIT;
  constexpr int LP     = L * NPTS;
  constexpr int OFF_LD = (CFG == 0) ? 192 : 64;  // padded widths of head GEMMs
  constexpr int AW_LD  = (CFG == 0) ? 128 : 64;

  int wid = blockIdx.x * (blockDim.x >> 5) + (threadIdx.x >> 5);
  int lane = threadIdx.x & 31;
  if (wid >= NB * Nq * NHEAD) return;
  int m = wid % NHEAD;
  int q = (wid / NHEAD) % Nq;
  int b = wid / (NHEAD * Nq);

  float rx, ry;
  if (CFG == 0) {
    rx = ((q & 31) + 0.5f) * (1.0f / 32.0f);
    ry = ((q >> 5) + 0.5f) * (1.0f / 32.0f);
  } else {
    if (q < 4096)      { int e = q;        rx = ((e & 63) + 0.5f) * (1.0f / 64.0f); ry = ((e >> 6) + 0.5f) * (1.0f / 64.0f); }
    else if (q < 5120) { int e = q - 4096; rx = ((e & 31) + 0.5f) * (1.0f / 32.0f); ry = ((e >> 5) + 0.5f) * (1.0f / 32.0f); }
    else               { int e = q - 5120; rx = ((e & 15) + 0.5f) * (1.0f / 16.0f); ry = ((e >> 4) + 0.5f) * (1.0f / 16.0f); }
  }

  // softmax over L*P attention logits (redundant per lane; <=12 values)
  const float* awp = AW + (size_t)(b * Nq + q) * AW_LD + m * LP;
  float lw[12];
  float mx = -1e30f;
#pragma unroll
  for (int j = 0; j < LP; j++) { lw[j] = awp[j]; mx = fmaxf(mx, lw[j]); }
  float sum = 0.0f;
#pragma unroll
  for (int j = 0; j < LP; j++) { lw[j] = __expf(lw[j] - mx); sum += lw[j]; }
  float inv = 1.0f / sum;

  int Hl[3], Wl[3], s0[3];
  if (CFG == 0) {
    Hl[0] = 64; Wl[0] = 64; s0[0] = 0;
    Hl[1] = 32; Wl[1] = 32; s0[1] = 4096;
    Hl[2] = 16; Wl[2] = 16; s0[2] = 5120;
  } else {
    Hl[0] = 32; Wl[0] = 32; s0[0] = 0;
  }

  const float* offp = OFF + (size_t)(b * Nq + q) * OFF_LD + m * (LP * 2);
  const float* vbase = V + (size_t)b * S * CDIM + m * DHEAD + lane * 4;

  float a0 = 0.f, a1 = 0.f, a2 = 0.f, a3 = 0.f;
  for (int l = 0; l < L; l++) {
    int H = Hl[l], W = Wl[l];
    float fW = (float)W, fH = (float)H;
#pragma unroll
    for (int p = 0; p < NPTS; p++) {
      float ox = offp[(l * NPTS + p) * 2];
      float oy = offp[(l * NPTS + p) * 2 + 1];
      float xf = (rx + ox / fW) * fW - 0.5f;
      float yf = (ry + oy / fH) * fH - 0.5f;
      float x0 = floorf(xf), y0 = floorf(yf);
      float wx = xf - x0, wy = yf - y0;
      int ix0 = (int)x0, iy0 = (int)y0;
      float aw = lw[l * NPTS + p] * inv;
#pragma unroll
      for (int cy = 0; cy < 2; cy++) {
#pragma unroll
        for (int cx = 0; cx < 2; cx++) {
          int xi = ix0 + cx, yi = iy0 + cy;
          if (xi >= 0 && xi < W && yi >= 0 && yi < H) {
            float wgt = (cx ? wx : 1.0f - wx) * (cy ? wy : 1.0f - wy) * aw;
            const float* vp = vbase + (size_t)(s0[l] + yi * W + xi) * CDIM;
            floatx4 f = *reinterpret_cast<const floatx4*>(vp);
            a0 += wgt * f[0]; a1 += wgt * f[1]; a2 += wgt * f[2]; a3 += wgt * f[3];
          }
        }
      }
    }
  }
  unsigned short* op = O + ((size_t)b * Nq + q) * CDIM + m * DHEAD + lane * 4;
  op[0] = f2bf(a0); op[1] = f2bf(a1); op[2] = f2bf(a2); op[3] = f2bf(a3);
}

// ---------------------------------------------------------------------------
// Depthwise 3x3 'SAME' conv per pyramid level + bias + exact GELU, output bf16.
// Input h: (B, 5376, 192) channel-last. One thread per (b, token, channel).
// ---------------------------------------------------------------------------
__global__ void dwconv_gelu_kernel(const float* __restrict__ Hin, const float* __restrict__ dw,
                                   const float* __restrict__ db, unsigned short* __restrict__ O) {
  int idx = blockIdx.x * 256 + threadIdx.x;
  if (idx >= NB * NPYR * 192) return;
  int ch = idx % 192;
  int n  = (idx / 192) % NPYR;
  int b  = idx / (192 * NPYR);
  int H, W, s0, y, x;
  if (n < 4096)      { H = 64; W = 64; s0 = 0;    int e = n;        y = e >> 6; x = e & 63; }
  else if (n < 5120) { H = 32; W = 32; s0 = 4096; int e = n - 4096; y = e >> 5; x = e & 31; }
  else               { H = 16; W = 16; s0 = 5120; int e = n - 5120; y = e >> 4; x = e & 15; }
  float acc = db[ch];
#pragma unroll
  for (int dy = 0; dy < 3; dy++) {
    int yy = y + dy - 1;
    if (yy < 0 || yy >= H) continue;
#pragma unroll
    for (int dx = 0; dx < 3; dx++) {
      int xx = x + dx - 1;
      if (xx < 0 || xx >= W) continue;
      acc += Hin[((size_t)b * NPYR + s0 + yy * W + xx) * 192 + ch] * dw[ch * 9 + dy * 3 + dx];
    }
  }
  float g = 0.5f * acc * (1.0f + erff(acc * 0.70710678118654752f));
  O[idx] = f2bf(g);
}

// ---------------------------------------------------------------------------
// Host launcher
// ---------------------------------------------------------------------------
static inline void launch_gemm(const unsigned short* A, const unsigned short* Wt,
                               const float* bias, const float* res, const float* gamma,
                               float* Out, int M, int Npad, int Nvalid, int K, int mode,
                               hipStream_t s) {
  dim3 g(Npad / 64, M / 256);
  gemm_bf16_wmma_kernel<<<g, 256, 0, s>>>(A, Wt, bias, res, gamma, Out, Npad, Nvalid, K, mode);
}

static inline void launch_packw(const float* W, unsigned short* Wt, int K, int N, int Npad,
                                hipStream_t s) {
  int tot = Npad * K;
  packw_kernel<<<(tot + 255) / 256, 256, 0, s>>>(W, Wt, K, N, Npad);
}

static inline void launch_ln(const float* X, const float* g, const float* b,
                             unsigned short* O, int Ntok, hipStream_t s) {
  ln_bf16_kernel<<<Ntok, 256, 0, s>>>(X, g, b, O);
}

extern "C" void kernel_launch(void* const* d_in, const int* in_sizes, int n_in,
                              void* d_out, int out_size, void* d_ws, size_t ws_size,
                              hipStream_t stream) {
  (void)in_sizes; (void)n_in; (void)out_size; (void)ws_size;
  const float* x         = (const float*)d_in[0];
  const float* c         = (const float*)d_in[1];
  const float* inj_n1g   = (const float*)d_in[2];
  const float* inj_n1b   = (const float*)d_in[3];
  const float* inj_n2g   = (const float*)d_in[4];
  const float* inj_n2b   = (const float*)d_in[5];
  const float* inj_off_w = (const float*)d_in[6];
  const float* inj_off_b = (const float*)d_in[7];
  const float* inj_aw_w  = (const float*)d_in[8];
  const float* inj_aw_b  = (const float*)d_in[9];
  const float* inj_val_w = (const float*)d_in[10];
  const float* inj_val_b = (const float*)d_in[11];
  const float* inj_out_w = (const float*)d_in[12];
  const float* inj_out_b = (const float*)d_in[13];
  const float* ext_n1g   = (const float*)d_in[14];
  const float* ext_n1b   = (const float*)d_in[15];
  const float* ext_n2g   = (const float*)d_in[16];
  const float* ext_n2b   = (const float*)d_in[17];
  const float* ext_off_w = (const float*)d_in[18];
  const float* ext_off_b = (const float*)d_in[19];
  const float* ext_aw_w  = (const float*)d_in[20];
  const float* ext_aw_b  = (const float*)d_in[21];
  const float* ext_val_w = (const float*)d_in[22];
  const float* ext_val_b = (const float*)d_in[23];
  const float* ext_out_w = (const float*)d_in[24];
  const float* ext_out_b = (const float*)d_in[25];
  const float* inj_gamma = (const float*)d_in[26];
  const float* ext_ng    = (const float*)d_in[27];
  const float* ext_nb    = (const float*)d_in[28];
  const float* ext_fc1_w = (const float*)d_in[29];
  const float* ext_fc1_b = (const float*)d_in[30];
  const float* ext_dw_w  = (const float*)d_in[31];
  const float* ext_dw_b  = (const float*)d_in[32];
  const float* ext_fc2_w = (const float*)d_in[33];
  const float* ext_fc2_b = (const float*)d_in[34];

  float* xout = (float*)d_out;                           // (4,1024,768)
  float* cout = xout + (size_t)NB * NVIT * CDIM;         // (4,5376,768)

  char* wsb = (char*)d_ws;
  size_t off = 0;
  auto take = [&](size_t bytes) -> char* {
    char* p = wsb + off;
    off = (off + bytes + 255) & ~(size_t)255;
    return p;
  };
  // padded weights (Npad x K, bf16)
  unsigned short* wt_inj_off = (unsigned short*)take((size_t)192 * 768 * 2);
  unsigned short* wt_inj_aw  = (unsigned short*)take((size_t)128 * 768 * 2);
  unsigned short* wt_inj_val = (unsigned short*)take((size_t)768 * 768 * 2);
  unsigned short* wt_inj_out = (unsigned short*)take((size_t)768 * 768 * 2);
  unsigned short* wt_ext_off = (unsigned short*)take((size_t)64 * 768 * 2);
  unsigned short* wt_ext_aw  = (unsigned short*)take((size_t)64 * 768 * 2);
  unsigned short* wt_ext_val = (unsigned short*)take((size_t)768 * 768 * 2);
  unsigned short* wt_ext_out = (unsigned short*)take((size_t)768 * 768 * 2);
  unsigned short* wt_fc1     = (unsigned short*)take((size_t)192 * 768 * 2);
  unsigned short* wt_fc2     = (unsigned short*)take((size_t)768 * 192 * 2);
  unsigned short* bf_big = (unsigned short*)take((size_t)NB * NPYR * CDIM * 2); // kvc / qc / ln2 / attn_e
  unsigned short* bf_sml = (unsigned short*)take((size_t)NB * NVIT * CDIM * 2); // qx / kx / attn_inj
  float* f32_big = (float*)take((size_t)NB * NPYR * CDIM * 4);                  // v_inj / h / h2
  float* f32_sml = (float*)take((size_t)NB * NVIT * CDIM * 4);                  // v_ext
  float* offsbuf = (float*)take((size_t)6 * 1024 * 1024);   // max(4096*192, 21504*64)*4
  float* awbuf   = (float*)take((size_t)6 * 1024 * 1024);   // max(4096*128, 21504*64)*4

  const int Mv = NB * NVIT;   // 4096
  const int Mp = NB * NPYR;   // 21504

  // -------- pack weights (f32 KxN -> bf16 NpadxK, zero padded) --------
  launch_packw(inj_off_w, wt_inj_off, 768, 144, 192, stream);
  launch_packw(inj_aw_w,  wt_inj_aw,  768,  72, 128, stream);
  launch_packw(inj_val_w, wt_inj_val, 768, 768, 768, stream);
  launch_packw(inj_out_w, wt_inj_out, 768, 768, 768, stream);
  launch_packw(ext_off_w, wt_ext_off, 768,  48,  64, stream);
  launch_packw(ext_aw_w,  wt_ext_aw,  768,  24,  64, stream);
  launch_packw(ext_val_w, wt_ext_val, 768, 768, 768, stream);
  launch_packw(ext_out_w, wt_ext_out, 768, 768, 768, stream);
  launch_packw(ext_fc1_w, wt_fc1,     768, 192, 192, stream);
  launch_packw(ext_fc2_w, wt_fc2,     192, 768, 768, stream);

  // ================= Injector =================
  launch_ln(x, inj_n1g, inj_n1b, bf_sml, Mv, stream);          // qx (bf16)
  launch_ln(c, inj_n2g, inj_n2b, bf_big, Mp, stream);          // kvc (bf16)
  launch_gemm(bf_big, wt_inj_val, inj_val_b, nullptr, nullptr,
              f32_big, Mp, 768, 768, 768, 0, stream);          // v_inj
  launch_gemm(bf_sml, wt_inj_off, inj_off_b, nullptr, nullptr,
              offsbuf, Mv, 192, 144, 768, 0, stream);          // sampling offsets
  launch_gemm(bf_sml, wt_inj_aw, inj_aw_b, nullptr, nullptr,
              awbuf, Mv, 128, 72, 768, 0, stream);             // attention logits
  {
    int waves = NB * NVIT * NHEAD;                             // 24576
    msda_kernel<0><<<waves / 8, 256, 0, stream>>>(f32_big, offsbuf, awbuf, bf_sml);
  }
  launch_gemm(bf_sml, wt_inj_out, inj_out_b, x, inj_gamma,
              xout, Mv, 768, 768, 768, 1, stream);             // x + gamma*proj(attn)

  // ================= Extractor =================
  launch_ln(c, ext_n1g, ext_n1b, bf_big, Mp, stream);          // qc (bf16)
  launch_ln(xout, ext_n2g, ext_n2b, bf_sml, Mv, stream);       // kx (bf16)
  launch_gemm(bf_sml, wt_ext_val, ext_val_b, nullptr, nullptr,
              f32_sml, Mv, 768, 768, 768, 0, stream);          // v_ext
  launch_gemm(bf_big, wt_ext_off, ext_off_b, nullptr, nullptr,
              offsbuf, Mp, 64, 48, 768, 0, stream);
  launch_gemm(bf_big, wt_ext_aw, ext_aw_b, nullptr, nullptr,
              awbuf, Mp, 64, 24, 768, 0, stream);
  {
    int waves = NB * NPYR * NHEAD;                             // 129024
    msda_kernel<1><<<waves / 8, 256, 0, stream>>>(f32_sml, offsbuf, awbuf, bf_big);
  }
  launch_gemm(bf_big, wt_ext_out, ext_out_b, c, nullptr,
              cout, Mp, 768, 768, 768, 2, stream);             // c + proj(attn)

  // ================= ConvFFN =================
  launch_ln(cout, ext_ng, ext_nb, bf_big, Mp, stream);         // ln(c) bf16
  launch_gemm(bf_big, wt_fc1, ext_fc1_b, nullptr, nullptr,
              f32_big, Mp, 192, 192, 768, 0, stream);          // h = fc1
  unsigned short* h2 = (unsigned short*)((char*)f32_big + (size_t)Mp * 192 * 4);
  {
    int tot = NB * NPYR * 192;
    dwconv_gelu_kernel<<<(tot + 255) / 256, 256, 0, stream>>>(f32_big, ext_dw_w, ext_dw_b, h2);
  }
  launch_gemm(h2, wt_fc2, ext_fc2_b, cout, nullptr,
              cout, Mp, 768, 768, 192, 2, stream);             // c += fc2(gelu(dw(h)))
}
